// CAM_41377714929724
// MI455X (gfx1250) — matrix operand order repaired and checked by reading, not compiled, not run on hardware
//
#include <hip/hip_runtime.h>
#include <hip/hip_bf16.h>

// ---------------------------------------------------------------------------
// CAM cross-attention on MI455X (gfx1250, wave32, WMMA).
//
//   a1  = f1 @ W                    (GEMM 1)
//   cc  = a1 @ f2^T                 (GEMM 2)
//   row stats:  rmax[s], rinv[s] = 1/sum_t exp(cc[s,t]-rmax)
//   col stats:  cmax[t], cinv[t] = 1/sum_s exp(cc[s,t]-cmax)   (chunked)
//   out2 = RowSoftmax(cc)   @ f2^T  (GEMM 3, exp fused into A-tile load)
//   out1 = RowSoftmax(cc^T) @ f1^T  (GEMM 4, exp fused into A-tile load)
//
// Precision: split-f16 (x = hi + lo) with fp32 accumulation:
//   x*y ~= hi_x*hi_y + hi_x*lo_y + lo_x*hi_y   -> ~2^-22 relative error.
//
// GEMM: 128x128 block tile, 256 threads = 8 waves (4x2), wave = 32x64
//       (24 WMMAs / wave / k-step), K-step 32, register prefetch of the
//       next global tile overlapping the WMMA phase.
// ---------------------------------------------------------------------------

typedef __attribute__((ext_vector_type(16))) _Float16 v16h;
typedef __attribute__((ext_vector_type(8)))  float    v8f;

#define LDSK 40            // LDS row stride in halves (80B: 16B-aligned, conflict-free)
#define MATSZ (1024ull*1024ull)

template<int AMODE, int BMODE, bool EXPA>
__launch_bounds__(256)
__global__ void gemm_kernel(const float* __restrict__ Abase, size_t aStride,
                            const float* __restrict__ Bbase, size_t bStride,
                            float* __restrict__ Cbase,
                            const float* __restrict__ stMax,
                            const float* __restrict__ stInv) {
  __shared__ _Float16 lds_a_hi[128 * LDSK];
  __shared__ _Float16 lds_a_lo[128 * LDSK];
  __shared__ _Float16 lds_b_hi[128 * LDSK];
  __shared__ _Float16 lds_b_lo[128 * LDSK];

  const int b  = blockIdx.z;
  const int m0 = blockIdx.y * 128;
  const int n0 = blockIdx.x * 128;
  const float* A = Abase + (size_t)b * aStride;
  const float* B = Bbase + (size_t)b * bStride;
  float*       C = Cbase + (size_t)b * MATSZ;

  const int tid  = threadIdx.x;
  const int lane = tid & 31;        // wave32
  const int w    = tid >> 5;        // wave id 0..7
  const int wm   = w >> 1;          // wave row (0..3) -> 32 rows each
  const int wn   = w & 1;           // wave col (0..1) -> 64 cols each
  const int lm   = lane & 15;       // row/col within 16x16 tile
  const int h    = lane >> 4;       // lane half (K-half selector)

  // ---- hoisted softmax stats (k-invariant per thread) ----
  float smx[4], siv[4];
  if (EXPA) {
    if (AMODE == 0) {
      #pragma unroll
      for (int ii = 0; ii < 4; ++ii) {
        const int r = (tid >> 3) + ii * 32;
        smx[ii] = stMax[(size_t)b * 1024 + m0 + r];
        siv[ii] = stInv[(size_t)b * 1024 + m0 + r];
      }
    } else {
      #pragma unroll
      for (int j = 0; j < 4; ++j) {
        const int m = ((tid & 31) << 2) + j;
        smx[j] = stMax[(size_t)b * 1024 + m0 + m];
        siv[j] = stInv[(size_t)b * 1024 + m0 + m];
      }
    }
  }

  float4 sa[4], sb[4];   // register-staged global tiles (prefetch)

  auto load_tiles = [&](int k0) {
    if (AMODE == 0) {                         // A[m][k], coalesced along k
      #pragma unroll
      for (int ii = 0; ii < 4; ++ii) {
        const int idx = tid + ii * 256;
        const int r = idx >> 3, kk = (idx & 7) << 2;
        sa[ii] = *(const float4*)(A + (size_t)(m0 + r) * 1024 + k0 + kk);
      }
    } else {                                  // A[m][k]=src[k][m], coalesced along m
      #pragma unroll
      for (int ii = 0; ii < 4; ++ii) {
        const int idx = tid + ii * 256;
        const int kr = idx >> 5, mm = (idx & 31) << 2;
        sa[ii] = *(const float4*)(A + (size_t)(k0 + kr) * 1024 + m0 + mm);
      }
    }
    if (BMODE == 0) {                         // B[k][n], coalesced along n
      #pragma unroll
      for (int ii = 0; ii < 4; ++ii) {
        const int idx = tid + ii * 256;
        const int kr = idx >> 5, nn = (idx & 31) << 2;
        sb[ii] = *(const float4*)(B + (size_t)(k0 + kr) * 1024 + n0 + nn);
      }
    } else {                                  // B[k][n]=src[n][k], coalesced along k
      #pragma unroll
      for (int ii = 0; ii < 4; ++ii) {
        const int idx = tid + ii * 256;
        const int r = idx >> 3, kk = (idx & 7) << 2;
        sb[ii] = *(const float4*)(B + (size_t)(n0 + r) * 1024 + k0 + kk);
      }
    }
  };

  auto store_lds = [&]() {
    if (AMODE == 0) {
      #pragma unroll
      for (int ii = 0; ii < 4; ++ii) {
        const int idx = tid + ii * 256;
        const int r = idx >> 3, kk = (idx & 7) << 2;
        const float* pv = (const float*)&sa[ii];
        #pragma unroll
        for (int j = 0; j < 4; ++j) {
          float x = pv[j];
          if (EXPA) x = __expf(x - smx[ii]) * siv[ii];
          const _Float16 hi = (_Float16)x;
          lds_a_hi[r * LDSK + kk + j] = hi;
          lds_a_lo[r * LDSK + kk + j] = (_Float16)(x - (float)hi);
        }
      }
    } else {
      #pragma unroll
      for (int ii = 0; ii < 4; ++ii) {
        const int idx = tid + ii * 256;
        const int kr = idx >> 5, mm = (idx & 31) << 2;
        const float* pv = (const float*)&sa[ii];
        #pragma unroll
        for (int j = 0; j < 4; ++j) {
          float x = pv[j];
          if (EXPA) x = __expf(x - smx[j]) * siv[j];
          const _Float16 hi = (_Float16)x;
          lds_a_hi[(mm + j) * LDSK + kr] = hi;
          lds_a_lo[(mm + j) * LDSK + kr] = (_Float16)(x - (float)hi);
        }
      }
    }
    if (BMODE == 0) {                         // transpose K x N -> [n][k] in LDS
      #pragma unroll
      for (int ii = 0; ii < 4; ++ii) {
        const int idx = tid + ii * 256;
        const int kr = idx >> 5, nn = (idx & 31) << 2;
        const float* pv = (const float*)&sb[ii];
        #pragma unroll
        for (int j = 0; j < 4; ++j) {
          const _Float16 hi = (_Float16)pv[j];
          lds_b_hi[(nn + j) * LDSK + kr] = hi;
          lds_b_lo[(nn + j) * LDSK + kr] = (_Float16)(pv[j] - (float)hi);
        }
      }
    } else {
      #pragma unroll
      for (int ii = 0; ii < 4; ++ii) {
        const int idx = tid + ii * 256;
        const int r = idx >> 3, kk = (idx & 7) << 2;
        const float* pv = (const float*)&sb[ii];
        #pragma unroll
        for (int j = 0; j < 4; ++j) {
          const _Float16 hi = (_Float16)pv[j];
          lds_b_hi[r * LDSK + kk + j] = hi;
          lds_b_lo[r * LDSK + kk + j] = (_Float16)(pv[j] - (float)hi);
        }
      }
    }
  };

  v8f acc[2][4] = {};

  load_tiles(0);
  for (int k0 = 0; k0 < 1024; k0 += 32) {
    store_lds();
    __syncthreads();
    if (k0 + 32 < 1024) load_tiles(k0 + 32);   // prefetch overlaps WMMA phase

    // ---- fragment loads per ISA 16-bit layouts ----
    // A 16x32: lane {M=lm, half h}: elems 0..7 <- K[h*8..+8], 8..15 <- K[16+h*8..+8]
    v16h ah[2], al[2];
    #pragma unroll
    for (int mi = 0; mi < 2; ++mi) {
      const int row = wm * 32 + mi * 16 + lm;
      const _Float16* ph = &lds_a_hi[row * LDSK];
      const _Float16* pl = &lds_a_lo[row * LDSK];
      #pragma unroll
      for (int e = 0; e < 8; ++e) {
        ah[mi][e]     = ph[h * 8 + e];
        ah[mi][8 + e] = ph[16 + h * 8 + e];
        al[mi][e]     = pl[h * 8 + e];
        al[mi][8 + e] = pl[16 + h * 8 + e];
      }
    }
    // B 32x16: lane {N=lm, half h}: elems 0..7 <- K[h*8..+8], 8..15 <- K[16+h*8..+8]
    #pragma unroll
    for (int ni = 0; ni < 4; ++ni) {
      const int col = wn * 64 + ni * 16 + lm;
      const _Float16* ph = &lds_b_hi[col * LDSK];
      const _Float16* pl = &lds_b_lo[col * LDSK];
      v16h bh, bl;
      #pragma unroll
      for (int e = 0; e < 16; ++e) { const int kq = (e < 8) ? (h * 8 + e) : (8 + h * 8 + e); bh[e] = ph[kq]; bl[e] = pl[kq]; }
      #pragma unroll
      for (int mi = 0; mi < 2; ++mi) {
        acc[mi][ni] = __builtin_amdgcn_wmma_f32_16x16x32_f16(
            false, ah[mi], false, bh, (short)0, acc[mi][ni], false, false);
        acc[mi][ni] = __builtin_amdgcn_wmma_f32_16x16x32_f16(
            false, ah[mi], false, bl, (short)0, acc[mi][ni], false, false);
        acc[mi][ni] = __builtin_amdgcn_wmma_f32_16x16x32_f16(
            false, al[mi], false, bh, (short)0, acc[mi][ni], false, false);
      }
    }
    __syncthreads();
  }

  // ---- store C: 16x16 f32 tile layout: N = lm, M = vgpr + 8*h ----
  #pragma unroll
  for (int mi = 0; mi < 2; ++mi)
    #pragma unroll
    for (int ni = 0; ni < 4; ++ni) {
      const int col  = n0 + wn * 64 + ni * 16 + lm;
      const int rowb = m0 + wm * 32 + mi * 16 + h * 8;
      #pragma unroll
      for (int j = 0; j < 8; ++j)
        C[(size_t)(rowb + j) * 1024 + col] = acc[mi][ni][j];
    }
}

// ---------------------------------------------------------------------------
// Row softmax stats: one block per (row, batch); 256 threads x float4 = 1024.
// ---------------------------------------------------------------------------
__global__ void row_stats_kernel(const float* __restrict__ cc,
                                 float* __restrict__ rmax, float* __restrict__ rinv) {
  const int row = blockIdx.x, b = blockIdx.y, tid = threadIdx.x;
  const float* p = cc + ((size_t)b * 1024 + row) * 1024;
  float4 v = ((const float4*)p)[tid];
  __shared__ float red[256];
  float m = fmaxf(fmaxf(v.x, v.y), fmaxf(v.z, v.w));
  red[tid] = m; __syncthreads();
  for (int s = 128; s > 0; s >>= 1) {
    if (tid < s) red[tid] = fmaxf(red[tid], red[tid + s]);
    __syncthreads();
  }
  m = red[0]; __syncthreads();
  float sum = __expf(v.x - m) + __expf(v.y - m) + __expf(v.z - m) + __expf(v.w - m);
  red[tid] = sum; __syncthreads();
  for (int s = 128; s > 0; s >>= 1) {
    if (tid < s) red[tid] += red[tid + s];
    __syncthreads();
  }
  if (tid == 0) {
    rmax[(size_t)b * 1024 + row] = m;
    rinv[(size_t)b * 1024 + row] = 1.0f / red[0];
  }
}

// ---------------------------------------------------------------------------
// Column softmax stats, chunked over the row dimension for parallelism:
// thread = column (consecutive lanes -> coalesced), 128 rows per chunk.
// ---------------------------------------------------------------------------
__global__ void col_stats_partial(const float* __restrict__ cc,
                                  float* __restrict__ pmax, float* __restrict__ psum) {
  const int t = blockIdx.x * 256 + threadIdx.x;   // column
  const int b = blockIdx.y;
  const int c = blockIdx.z;                        // row chunk 0..7
  const float* p = cc + (size_t)b * MATSZ + (size_t)c * 128 * 1024 + t;
  float m = -__builtin_inff(), s = 0.0f;
  for (int k = 0; k < 128; ++k) {
    float v = p[(size_t)k * 1024];
    if (v > m) { s = s * __expf(m - v) + 1.0f; m = v; }
    else       { s += __expf(v - m); }
  }
  const size_t o = ((size_t)c * 16 + b) * 1024 + t;
  pmax[o] = m;
  psum[o] = s;
}

__global__ void col_stats_combine(const float* __restrict__ pmax,
                                  const float* __restrict__ psum,
                                  float* __restrict__ cmax, float* __restrict__ cinv) {
  const int t = blockIdx.x * 256 + threadIdx.x;
  const int b = blockIdx.y;
  float m = -__builtin_inff();
  #pragma unroll
  for (int c = 0; c < 8; ++c)
    m = fmaxf(m, pmax[((size_t)c * 16 + b) * 1024 + t]);
  float s = 0.0f;
  #pragma unroll
  for (int c = 0; c < 8; ++c) {
    const size_t o = ((size_t)c * 16 + b) * 1024 + t;
    s += psum[o] * __expf(pmax[o] - m);
  }
  cmax[(size_t)b * 1024 + t] = m;
  cinv[(size_t)b * 1024 + t] = 1.0f / s;
}

// ---------------------------------------------------------------------------
extern "C" void kernel_launch(void* const* d_in, const int* in_sizes, int n_in,
                              void* d_out, int out_size, void* d_ws, size_t ws_size,
                              hipStream_t stream) {
  const float* f1 = (const float*)d_in[0];   // [16,1024,1024]
  const float* f2 = (const float*)d_in[1];   // [16,1024,1024]
  const float* W  = (const float*)d_in[2];   // [1024,1024]
  float* out = (float*)d_out;
  float* ws  = (float*)d_ws;

  float* a1   = ws;                          // 16 * 1M floats
  float* cc   = ws + 16 * MATSZ;             // 16 * 1M floats
  float* rmax = ws + 32 * MATSZ;             // 16K each below
  float* rinv = rmax + 16 * 1024;
  float* cmax = rinv + 16 * 1024;
  float* cinv = cmax + 16 * 1024;
  float* pmax = cinv + 16 * 1024;            // 8 * 16K partials
  float* psum = pmax + 8 * 16 * 1024;

  float* out1 = out;                         // swapped atten_audiofeatures
  float* out2 = out + 16 * MATSZ;            // swapped atten_visualfeatures

  dim3 g(8, 8, 16), blk(256);

  // a1 = f1 @ W            (A plain batched, B plain shared)
  gemm_kernel<0, 0, false><<<g, blk, 0, stream>>>(f1, MATSZ, W, 0, a1, nullptr, nullptr);
  // cc = a1 @ f2^T         (A plain, B transposed)
  gemm_kernel<0, 1, false><<<g, blk, 0, stream>>>(a1, MATSZ, f2, MATSZ, cc, nullptr, nullptr);
  // softmax statistics
  row_stats_kernel<<<dim3(1024, 16), 256, 0, stream>>>(cc, rmax, rinv);
  col_stats_partial<<<dim3(4, 16, 8), 256, 0, stream>>>(cc, pmax, psum);
  col_stats_combine<<<dim3(4, 16),    256, 0, stream>>>(pmax, psum, cmax, cinv);
  // out2 = RowSoftmax(cc)   @ f2^T   (exp fused into A load)
  gemm_kernel<0, 1, true><<<g, blk, 0, stream>>>(cc, MATSZ, f2, MATSZ, out2, rmax, rinv);
  // out1 = RowSoftmax(cc^T) @ f1^T   (A transposed read of cc, col stats)
  gemm_kernel<1, 1, true><<<g, blk, 0, stream>>>(cc, MATSZ, f1, MATSZ, out1, cmax, cinv);
}
